// EncoderLayerWithAttn_27204322852992
// MI455X (gfx1250) — matrix operand order repaired and checked by reading, not compiled, not run on hardware
//
#include <hip/hip_runtime.h>
#include <hip/hip_bf16.h>

// ---------------- CDNA5 WMMA helpers ----------------
typedef __attribute__((ext_vector_type(16))) _Float16 v16h;
typedef __attribute__((ext_vector_type(8)))  float    v8f;

#define BD (4)        // batch
#define SD (1024)     // seq
#define DD (1024)     // d_model
#define HD (16)       // heads
#define DH (64)       // head dim
#define FD (4096)     // dff

__device__ __forceinline__ v8f wmma_f16(v16h a, v16h b, v8f c) {
    return __builtin_amdgcn_wmma_f32_16x16x32_f16(false, a, false, b, (short)0, c, false, false);
}

union frag_u { v16h v; uint4 u[2]; _Float16 h[16]; };

// A-fragment (16x32 f16): lane L -> row = L&15, group g = L>>4.
// halves j<8 : K = 8g + j ; j>=8 : K = 16 + 8g + (j-8)  => two contiguous 8-half runs.
__device__ __forceinline__ v16h load_a_frag(const _Float16* __restrict__ rowptr, int g) {
    frag_u x;
    x.u[0] = *(const uint4*)(rowptr + 8 * g);
    x.u[1] = *(const uint4*)(rowptr + 16 + 8 * g);
    return x.v;
}
// B-fragment (32x16 f16): lane L -> col = L&15, group g = L>>4.
// halves j<8 : K = 8g + j ; j>=8 : K = 16 + 8g + (j-8)  => two contiguous 8-half runs.
__device__ __forceinline__ v16h load_b_frag(const _Float16* __restrict__ colptr, int g) {
    frag_u x;
    x.u[0] = *(const uint4*)(colptr + 8 * g);
    x.u[1] = *(const uint4*)(colptr + 16 + 8 * g);
    return x.v;
}

// ---------------- f32 -> f16 convert ----------------
__global__ void cvt_f16_kernel(const float* __restrict__ in, _Float16* __restrict__ out, int n) {
    int i = blockIdx.x * blockDim.x + threadIdx.x;
    if (i < n) out[i] = (_Float16)in[i];
}

// ---------------- V transpose: qkv16 v-slice -> vT[B,H,DH,S] ----------------
__global__ __launch_bounds__(256) void vtrans_kernel(const _Float16* __restrict__ qkv,
                                                     _Float16* __restrict__ vT) {
    const int bs = blockIdx.x;              // b*S + s
    const int b  = bs >> 10;
    const int s  = bs & 1023;
    const _Float16* vrow = qkv + (size_t)bs * (3 * DD) + 2 * DD;
    #pragma unroll
    for (int i = 0; i < 4; ++i) {
        const int d  = threadIdx.x + 256 * i;   // 0..1023
        const int h  = d >> 6;
        const int dh = d & 63;
        vT[(((size_t)(b * HD + h) * DH) + dh) * SD + s] = vrow[d];
    }
}

// ------- blocked WMMA GEMM: Y = A * W^T + bias (+res)(relu); wave tile 32x32 -------
// A: [M,K] f16 (ld=lda), W: [N,K] f16 (ld=ldw), out ld=ldo.
// Block = 128 threads = 4 waves in 2x2 -> 64x64 block tile. 4 WMMA per K-step per wave.
template <bool RELU, bool OUT16, bool RES>
__global__ __launch_bounds__(128) void gemm32_kernel(
    const _Float16* __restrict__ A, const _Float16* __restrict__ W,
    const float* __restrict__ bias, const float* __restrict__ res,
    float* __restrict__ outF, _Float16* __restrict__ outH,
    int K, int lda, int ldw, int ldo)
{
    const int lane = threadIdx.x & 31;
    const int wid  = threadIdx.x >> 5;
    const int g    = lane >> 4;
    const int l15  = lane & 15;
    const int m0   = blockIdx.y * 64 + (wid >> 1) * 32;
    const int n0   = blockIdx.x * 64 + (wid & 1) * 32;

    const _Float16* arow0 = A + (size_t)(m0 + l15) * lda;
    const _Float16* arow1 = arow0 + (size_t)16 * lda;
    const _Float16* wrow0 = W + (size_t)(n0 + l15) * ldw;
    const _Float16* wrow1 = wrow0 + (size_t)16 * ldw;

    v8f c00 = {}, c01 = {}, c10 = {}, c11 = {};
    for (int k0 = 0; k0 < K; k0 += 32) {
        const v16h a0 = load_a_frag(arow0 + k0, g);
        const v16h a1 = load_a_frag(arow1 + k0, g);
        const v16h b0 = load_b_frag(wrow0 + k0, g);
        const v16h b1 = load_b_frag(wrow1 + k0, g);
        c00 = wmma_f16(a0, b0, c00);
        c01 = wmma_f16(a0, b1, c01);
        c10 = wmma_f16(a1, b0, c10);
        c11 = wmma_f16(a1, b1, c11);
    }

    const int na = n0 + l15;
    const int nb = na + 16;
    const float bva = bias[na];
    const float bvb = bias[nb];
    #pragma unroll
    for (int r = 0; r < 8; ++r) {
        const int ma = m0 + 8 * g + r;
        const int mb = ma + 16;
        float v00 = c00[r] + bva, v01 = c01[r] + bvb;
        float v10 = c10[r] + bva, v11 = c11[r] + bvb;
        if (RES) {
            v00 += res[(size_t)ma * ldo + na]; v01 += res[(size_t)ma * ldo + nb];
            v10 += res[(size_t)mb * ldo + na]; v11 += res[(size_t)mb * ldo + nb];
        }
        if (RELU) {
            v00 = fmaxf(v00, 0.0f); v01 = fmaxf(v01, 0.0f);
            v10 = fmaxf(v10, 0.0f); v11 = fmaxf(v11, 0.0f);
        }
        if (OUT16) {
            outH[(size_t)ma * ldo + na] = (_Float16)v00;
            outH[(size_t)ma * ldo + nb] = (_Float16)v01;
            outH[(size_t)mb * ldo + na] = (_Float16)v10;
            outH[(size_t)mb * ldo + nb] = (_Float16)v11;
        } else {
            outF[(size_t)ma * ldo + na] = v00;
            outF[(size_t)ma * ldo + nb] = v01;
            outF[(size_t)mb * ldo + na] = v10;
            outF[(size_t)mb * ldo + nb] = v11;
        }
    }
}

// ---------------- fused attention: scores -> softmax -> attn out + ctx ----------------
// qkv16: [B*S, 3*D] f16 (q at 0, k at 1024, v at 2048). vT: [B,H,DH,S] f16.
__global__ __launch_bounds__(128) void attn_kernel(
    const _Float16* __restrict__ qkv,
    const _Float16* __restrict__ vT,
    float* __restrict__ attnw,           // [B,H,S,S]
    _Float16* __restrict__ ctx)          // [B*S, D]
{
    __shared__ float sc[16 * SD];                        // 64 KB scores / exp
    __shared__ __align__(16) _Float16 pb[16 * SD];       // 32 KB probs f16
    __shared__ float red[16 * 8];
    __shared__ float rmax[16];
    __shared__ float rinv[16];

    const int b  = blockIdx.z;
    const int h  = blockIdx.y;
    const int q0 = blockIdx.x * 16;
    const int lane = threadIdx.x & 31;
    const int wid  = threadIdx.x >> 5;
    const int g    = lane >> 4;
    const int l15  = lane & 15;

    // ---- Q fragments (2 k-steps of 32 over DH=64), scaled by 1/sqrt(64) ----
    v16h aq[2];
    {
        const _Float16* qrow = qkv + ((size_t)b * SD + q0 + l15) * (3 * DD) + h * DH;
        #pragma unroll
        for (int ks = 0; ks < 2; ++ks) {
            frag_u x;
            x.u[0] = *(const uint4*)(qrow + ks * 32 + 8 * g);
            x.u[1] = *(const uint4*)(qrow + ks * 32 + 16 + 8 * g);
            #pragma unroll
            for (int j = 0; j < 16; ++j) x.h[j] = x.h[j] * (_Float16)0.125f;
            aq[ks] = x.v;
        }
    }

    // ---- scores: Q[16x64] x K^T[64x1024] ; wave handles 16 column tiles ----
    for (int i = 0; i < 16; ++i) {
        const int ct = wid * 16 + i;
        const int n  = ct * 16 + l15;                    // key position
        const _Float16* kp = qkv + ((size_t)b * SD + n) * (3 * DD) + DD + h * DH;
        v8f c = {};
        #pragma unroll
        for (int ks = 0; ks < 2; ++ks) {
            frag_u x;
            x.u[0] = *(const uint4*)(kp + ks * 32 + 8 * g);
            x.u[1] = *(const uint4*)(kp + ks * 32 + 16 + 8 * g);
            c = wmma_f16(aq[ks], x.v, c);
        }
        #pragma unroll
        for (int r = 0; r < 8; ++r) sc[(8 * g + r) * SD + n] = c[r];
    }
    __syncthreads();

    // ---- softmax over 1024 keys per row; 8 threads per row ----
    const int row = threadIdx.x >> 3;
    const int seg = threadIdx.x & 7;
    float mx = -1e30f;
    for (int cc = 0; cc < 128; ++cc) mx = fmaxf(mx, sc[row * SD + seg * 128 + cc]);
    red[row * 8 + seg] = mx;
    __syncthreads();
    if (threadIdx.x < 16) {
        float m2 = red[threadIdx.x * 8];
        #pragma unroll
        for (int j = 1; j < 8; ++j) m2 = fmaxf(m2, red[threadIdx.x * 8 + j]);
        rmax[threadIdx.x] = m2;
    }
    __syncthreads();
    const float rm = rmax[row];
    float sum = 0.0f;
    for (int cc = 0; cc < 128; ++cc) {
        const int c_ = row * SD + seg * 128 + cc;
        const float e = __expf(sc[c_] - rm);
        sc[c_] = e;
        sum += e;
    }
    red[row * 8 + seg] = sum;
    __syncthreads();
    if (threadIdx.x < 16) {
        float s2 = 0.0f;
        #pragma unroll
        for (int j = 0; j < 8; ++j) s2 += red[threadIdx.x * 8 + j];
        rinv[threadIdx.x] = 1.0f / s2;
    }
    __syncthreads();
    const float inv = rinv[row];
    float* aw = attnw + (((size_t)(b * HD + h) * SD) + q0 + row) * SD;
    for (int cc = 0; cc < 128; ++cc) {
        const int c_ = seg * 128 + cc;
        const float p = sc[row * SD + c_] * inv;
        pb[row * SD + c_] = (_Float16)p;
        aw[c_] = p;
    }
    __syncthreads();

    // ---- ctx = P[16x1024] x V[1024x64]; wave wid owns dh-tile wid*16 ----
    {
        const int n = wid * 16 + l15;                    // dh index
        const _Float16* vcol = vT + (((size_t)(b * HD + h) * DH) + n) * SD;
        v8f c = {};
        for (int k0 = 0; k0 < SD; k0 += 32) {
            frag_u xa;
            const _Float16* pr = pb + l15 * SD + k0;     // LDS -> ds_load_b128
            xa.u[0] = *(const uint4*)(pr + 8 * g);
            xa.u[1] = *(const uint4*)(pr + 16 + 8 * g);
            frag_u xb;                                   // V^T: contiguous along seq
            xb.u[0] = *(const uint4*)(vcol + k0 + 8 * g);
            xb.u[1] = *(const uint4*)(vcol + k0 + 16 + 8 * g);
            c = wmma_f16(xa.v, xb.v, c);
        }
        #pragma unroll
        for (int r = 0; r < 8; ++r) {
            const int m = 8 * g + r;
            ctx[((size_t)b * SD + q0 + m) * DD + h * DH + n] = (_Float16)c[r];
        }
    }
}

// ---------------- row-wise add+LayerNorm (block per row, D=1024) ----------------
template <bool EMITH>
__global__ __launch_bounds__(256) void add_ln_kernel(
    const float* __restrict__ y, const float* __restrict__ gw, const float* __restrict__ bw,
    float* __restrict__ xo, _Float16* __restrict__ xh)
{
    __shared__ float sm[256];
    const size_t row = blockIdx.x;
    const float* yr = y + row * DD;
    float vals[4];
    float s = 0.0f;
    #pragma unroll
    for (int i = 0; i < 4; ++i) { vals[i] = yr[threadIdx.x + 256 * i]; s += vals[i]; }
    sm[threadIdx.x] = s; __syncthreads();
    for (int off = 128; off > 0; off >>= 1) {
        if (threadIdx.x < off) sm[threadIdx.x] += sm[threadIdx.x + off];
        __syncthreads();
    }
    const float mu = sm[0] * (1.0f / DD);
    __syncthreads();
    float v = 0.0f;
    #pragma unroll
    for (int i = 0; i < 4; ++i) { const float d = vals[i] - mu; v += d * d; }
    sm[threadIdx.x] = v; __syncthreads();
    for (int off = 128; off > 0; off >>= 1) {
        if (threadIdx.x < off) sm[threadIdx.x] += sm[threadIdx.x + off];
        __syncthreads();
    }
    const float rstd = rsqrtf(sm[0] * (1.0f / DD) + 1e-5f);
    #pragma unroll
    for (int i = 0; i < 4; ++i) {
        const int c = threadIdx.x + 256 * i;
        const float o = (vals[i] - mu) * rstd * gw[c] + bw[c];
        xo[row * DD + c] = o;
        if (EMITH) xh[row * DD + c] = (_Float16)o;
    }
}

// ---------------- host side ----------------
extern "C" void kernel_launch(void* const* d_in, const int* in_sizes, int n_in,
                              void* d_out, int out_size, void* d_ws, size_t ws_size,
                              hipStream_t stream) {
    const float* src   = (const float*)d_in[0];
    const float* w_in  = (const float*)d_in[1];
    const float* b_in  = (const float*)d_in[2];
    const float* w_out = (const float*)d_in[3];
    const float* b_out = (const float*)d_in[4];
    const float* w1    = (const float*)d_in[5];
    const float* b1    = (const float*)d_in[6];
    const float* w2    = (const float*)d_in[7];
    const float* b2    = (const float*)d_in[8];
    const float* g1    = (const float*)d_in[9];
    const float* be1   = (const float*)d_in[10];
    const float* g2    = (const float*)d_in[11];
    const float* be2   = (const float*)d_in[12];

    const size_t R = (size_t)BD * SD;               // 4096 rows
    float* outLN = (float*)d_out;                   // [B,S,D]
    float* attnW = (float*)d_out + R * DD;          // [B,H,S,S]

    // workspace carve-up
    char* p = (char*)d_ws;
    _Float16* src16  = (_Float16*)p; p += R * DD * 2;
    _Float16* w_in16 = (_Float16*)p; p += (size_t)3 * DD * DD * 2;
    _Float16* w_ot16 = (_Float16*)p; p += (size_t)DD * DD * 2;
    _Float16* w1h    = (_Float16*)p; p += (size_t)FD * DD * 2;
    _Float16* w2h    = (_Float16*)p; p += (size_t)DD * FD * 2;
    _Float16* qkv16  = (_Float16*)p; p += R * 3 * DD * 2;
    _Float16* vT     = (_Float16*)p; p += R * DD * 2;
    _Float16* ctx16  = (_Float16*)p; p += R * DD * 2;
    float*    y1     = (float*)p;    p += R * DD * 4;   // reused for y2
    float*    x      = (float*)p;    p += R * DD * 4;
    _Float16* x16    = (_Float16*)p; p += R * DD * 2;
    _Float16* ff1h   = (_Float16*)p; p += R * FD * 2;

    // 1) convert inputs/weights to f16
    cvt_f16_kernel<<<(R * DD) / 256, 256, 0, stream>>>(src,   src16,  (int)(R * DD));
    cvt_f16_kernel<<<(3 * DD * DD) / 256, 256, 0, stream>>>(w_in,  w_in16, 3 * DD * DD);
    cvt_f16_kernel<<<(DD * DD) / 256, 256, 0, stream>>>(w_out, w_ot16, DD * DD);
    cvt_f16_kernel<<<(FD * DD) / 256, 256, 0, stream>>>(w1,    w1h,    FD * DD);
    cvt_f16_kernel<<<(DD * FD) / 256, 256, 0, stream>>>(w2,    w2h,    DD * FD);

    // 2) qkv = src @ w_in^T + b_in  -> f16 [4096,3072]
    gemm32_kernel<false, true, false><<<dim3((3 * DD) / 64, R / 64), 128, 0, stream>>>(
        src16, w_in16, b_in, nullptr, nullptr, qkv16, DD, DD, DD, 3 * DD);

    // 2b) vT[B,H,DH,S] from qkv v-slice
    vtrans_kernel<<<R, 256, 0, stream>>>(qkv16, vT);

    // 3) fused attention (scores, softmax -> attnW, ctx)
    attn_kernel<<<dim3(SD / 16, HD, BD), 128, 0, stream>>>(qkv16, vT, attnW, ctx16);

    // 4) y1 = src + ctx @ w_out^T + b_out  (f32)
    gemm32_kernel<false, false, true><<<dim3(DD / 64, R / 64), 128, 0, stream>>>(
        ctx16, w_ot16, b_out, src, y1, nullptr, DD, DD, DD, DD);

    // 5) x = LN1(y1) (f32 + f16 copy)
    add_ln_kernel<true><<<R, 256, 0, stream>>>(y1, g1, be1, x, x16);

    // 6) ff1 = relu(x @ w1^T + b1) -> f16 [4096,4096]
    gemm32_kernel<true, true, false><<<dim3(FD / 64, R / 64), 128, 0, stream>>>(
        x16, w1h, b1, nullptr, nullptr, ff1h, DD, DD, DD, FD);

    // 7) y2 = x + ff1 @ w2^T + b2  (f32, reuse y1)
    gemm32_kernel<false, false, true><<<dim3(DD / 64, R / 64), 128, 0, stream>>>(
        ff1h, w2h, b2, x, y1, nullptr, FD, FD, FD, DD);

    // 8) out = LN2(y2)
    add_ln_kernel<false><<<R, 256, 0, stream>>>(y1, g2, be2, outLN, nullptr);
}
